// MSDeformAttnPixelDecoder_17308718203126
// MI455X (gfx1250) — compile-verified, matrix operand order checked by reading, not run on hardware
//
#include <hip/hip_runtime.h>
#include <hip/hip_bf16.h>

// ---------------- problem constants ----------------
constexpr int D   = 256;
constexpr int NH  = 8;
constexpr int DH  = 32;
constexpr int NL  = 3;
constexpr int NP  = 4;
constexpr int NLAYERS = 6;
constexpr int DFF = 1024;
constexpr int BB  = 4;                       // batch
constexpr int Q   = 64*64 + 32*32 + 16*16;   // 5376
constexpr int M   = BB * Q;                  // 21504 rows
constexpr int NATT = 96;                     // real attn logits
constexpr int NATTP = 128;                   // padded to multiple of 64
static_assert(M % 256 == 0, "M tile");

typedef __attribute__((ext_vector_type(16))) __bf16 v16bf;
typedef __attribute__((ext_vector_type(8)))  float  v8f;

__device__ __forceinline__ unsigned short f2bf(float f) {
  unsigned int u = __float_as_uint(f);
  u += 0x7FFFu + ((u >> 16) & 1u);           // round-to-nearest-even
  return (unsigned short)(u >> 16);
}

// ---- CDNA5 async global->LDS copy (no VGPR data path; tracked by ASYNCcnt) ----
__device__ __forceinline__ void async_ld_b128(unsigned lds_addr, const void* gaddr) {
  asm volatile("global_load_async_to_lds_b128 %0, %1, off"
               :: "v"(lds_addr), "v"(gaddr)
               : "memory");
}
__device__ __forceinline__ void wait_async0() {
  asm volatile("s_wait_asynccnt 0" ::: "memory");
}

// ---------------- flatten: [B,D,h,w] levels -> x[M,D], pos[M,D] ----------------
__global__ __launch_bounds__(256) void flatten_kernel(
    const float* __restrict__ s0, const float* __restrict__ p0,
    const float* __restrict__ s1, const float* __restrict__ p1,
    const float* __restrict__ s2, const float* __restrict__ p2,
    const float* __restrict__ lvl, float* __restrict__ x, float* __restrict__ pos)
{
  int idx = blockIdx.x * 256 + threadIdx.x;
  if (idx >= M * D) return;
  int ch = idx % D;
  int t  = idx / D;
  int q  = t % Q;
  int b  = t / Q;
  int l, hw, qi;
  const float *sp, *pp;
  if (q < 4096)      { l = 0; hw = 64*64; qi = q;        sp = s0; pp = p0; }
  else if (q < 5120) { l = 1; hw = 32*32; qi = q - 4096; sp = s1; pp = p1; }
  else               { l = 2; hw = 16*16; qi = q - 5120; sp = s2; pp = p2; }
  size_t si = (size_t)(b * D + ch) * hw + qi;
  x[idx]   = sp[si];
  pos[idx] = pp[si] + lvl[l * D + ch];
}

// ---------------- elementwise converters ----------------
__global__ __launch_bounds__(256) void cvt_f32_to_bf16(
    const float* __restrict__ in, unsigned short* __restrict__ out, int n)
{
  int i = blockIdx.x * 256 + threadIdx.x;
  if (i < n) out[i] = f2bf(in[i]);
}

// weights [L,K,N] f32 -> transposed bf16 [L,Npad,K], zero-padded rows n>=N
__global__ __launch_bounds__(256) void cvt_w_transpose(
    const float* __restrict__ in, unsigned short* __restrict__ out,
    int K, int N, int Npad, int L)
{
  int idx = blockIdx.x * 256 + threadIdx.x;
  if (idx >= L * Npad * K) return;
  int k = idx % K;
  int rem = idx / K;
  int n = rem % Npad;
  int l = rem / Npad;
  float v = (n < N) ? in[((size_t)l * K + k) * N + n] : 0.0f;
  out[idx] = f2bf(v);
}

// bias [L,96] -> [L,128] zero padded
__global__ __launch_bounds__(256) void pad_bias_attn(
    const float* __restrict__ in, float* __restrict__ out)
{
  int idx = blockIdx.x * 256 + threadIdx.x;
  if (idx >= NLAYERS * NATTP) return;
  int n = idx % NATTP, l = idx / NATTP;
  out[idx] = (n < NATT) ? in[l * NATT + n] : 0.0f;
}

__global__ __launch_bounds__(256) void prep_qx(
    const float* __restrict__ x, const float* __restrict__ pos,
    unsigned short* __restrict__ qb, unsigned short* __restrict__ xb, int n)
{
  int i = blockIdx.x * 256 + threadIdx.x;
  if (i < n) {
    float xv = x[i];
    qb[i] = f2bf(xv + pos[i]);
    xb[i] = f2bf(xv);
  }
}

// ---------------- WMMA bf16 GEMM:  C[M,N] = A[M,K] * Bt[N,K]^T + bias ----------------
// block = 256 threads = 8 waves; block tile 256x64; wave tile 32x64 (8 wmma/step);
// K step 32; double-buffered LDS fed by async global->LDS b128 copies (ASYNCcnt),
// one barrier per step; fragments via ds_load_b128.
constexpr int GBM = 256, GBN = 64, GBK = 32;
constexpr int GLDT = 40;   // ushort row stride: 80B, 16B-aligned rows

template <bool RELU, bool OUTBF16>
__global__ __launch_bounds__(256) void gemm_wmma(
    const unsigned short* __restrict__ A,    // [Mrows,K] bf16 bits
    const unsigned short* __restrict__ Bt,   // [N,K]     bf16 bits (pre-transposed)
    const float* __restrict__ bias,          // [N]
    float* __restrict__ Cf, unsigned short* __restrict__ Cb,
    int Mrows, int N, int K)
{
  __shared__ __align__(16) unsigned short As[2][GBM * GLDT];   // 2 x 20KB
  __shared__ __align__(16) unsigned short Bs[2][GBN * GLDT];   // 2 x  5KB

  const int tid   = threadIdx.x;
  const int wid   = tid >> 5;
  const int lane  = tid & 31;
  const int bm    = blockIdx.x * GBM;
  const int bn    = blockIdx.y * GBN;
  const int lhalf = lane >> 4;      // 0/1
  const int lm    = lane & 15;
  const int kh    = lhalf * 8;      // lanes 0-15: K {0..7,16..23}; 16-31: {8..15,24..31}
  const int waveRow = wid * 32;     // each wave owns 32 rows x all 64 cols

  // per-thread tile chunk: one b128 (8 bf16) per copy
  const int ar = tid >> 2, ac = (tid & 3) * 8;   // A rows ar + j*64, j=0..3
  const int br = tid >> 2, bc = (tid & 3) * 8;   // B rows 0..63

  const unsigned short* Aptr = A  + (size_t)bm * K;
  const unsigned short* Bptr = Bt + (size_t)bn * K;

  // loop-invariant per-thread source pointers and LDS byte addresses (stage 0)
  const unsigned short* ag[4];
#pragma unroll
  for (int j = 0; j < 4; ++j) ag[j] = Aptr + (size_t)(ar + j * 64) * K + ac;
  const unsigned short* bg = Bptr + (size_t)br * K + bc;

  unsigned aoff[4];
#pragma unroll
  for (int j = 0; j < 4; ++j)
    aoff[j] = (unsigned)(size_t)&As[0][(ar + j * 64) * GLDT + ac];
  const unsigned boff = (unsigned)(size_t)&Bs[0][br * GLDT + bc];
  constexpr unsigned ASTAGE = (unsigned)(GBM * GLDT * 2);   // 20480 B
  constexpr unsigned BSTAGE = (unsigned)(GBN * GLDT * 2);   //  5120 B

  auto issue_stage = [&](int s, int k0) {
#pragma unroll
    for (int j = 0; j < 4; ++j)
      async_ld_b128(aoff[j] + (unsigned)s * ASTAGE, ag[j] + k0);
    async_ld_b128(boff + (unsigned)s * BSTAGE, bg + k0);
  };

  v8f acc[2][4] = {};

  issue_stage(0, 0);
  wait_async0();
  __syncthreads();

  const int nsteps = K / GBK;
  int s = 0;
  for (int step = 0; step < nsteps; ++step) {
    if (step + 1 < nsteps) issue_stage(s ^ 1, (step + 1) * GBK);

    union Frag { v16bf v; uint4 q[2]; } af[2], bfr[4];
#pragma unroll
    for (int i = 0; i < 2; ++i) {
      const unsigned short* ap = &As[s][(waveRow + i * 16 + lm) * GLDT];
      af[i].q[0] = *(const uint4*)(ap + kh);
      af[i].q[1] = *(const uint4*)(ap + 16 + kh);
    }
#pragma unroll
    for (int j = 0; j < 4; ++j) {
      const unsigned short* bp = &Bs[s][(j * 16 + lm) * GLDT];
      bfr[j].q[0] = *(const uint4*)(bp + kh);
      bfr[j].q[1] = *(const uint4*)(bp + 16 + kh);
    }
#pragma unroll
    for (int i = 0; i < 2; ++i)
#pragma unroll
      for (int j = 0; j < 4; ++j)
        acc[i][j] = __builtin_amdgcn_wmma_f32_16x16x32_bf16(
            false, af[i].v, false, bfr[j].v, (short)0, acc[i][j], false, false);

    if (step + 1 < nsteps) wait_async0();   // next-stage LDS writes landed
    __syncthreads();
    s ^= 1;
  }

  // epilogue: tile (i,j); lane holds rows lhalf*8+r, col lm
#pragma unroll
  for (int i = 0; i < 2; ++i) {
    const int rowBase = bm + waveRow + i * 16 + lhalf * 8;
#pragma unroll
    for (int j = 0; j < 4; ++j) {
      const int col = bn + j * 16 + lm;
      const float bv = bias[col];
#pragma unroll
      for (int r = 0; r < 8; ++r) {
        float v = acc[i][j][r] + bv;
        if (RELU) v = fmaxf(v, 0.0f);
        size_t idx = (size_t)(rowBase + r) * N + col;
        if (OUTBF16) Cb[idx] = f2bf(v);
        else         Cf[idx] = v;
      }
    }
  }
}

// ---------------- softmax over NL*NP=12 per (row, head); buffer stride 128 ----------------
__global__ __launch_bounds__(256) void softmax12(float* __restrict__ aw)
{
  int t = blockIdx.x * 256 + threadIdx.x;
  if (t >= M * NH) return;
  int h = t % NH;
  int row = t / NH;
  float* p = aw + (size_t)row * NATTP + h * (NL * NP);
  float mx = p[0];
#pragma unroll
  for (int j = 1; j < 12; ++j) mx = fmaxf(mx, p[j]);
  float e[12], s = 0.f;
#pragma unroll
  for (int j = 0; j < 12; ++j) { e[j] = __expf(p[j] - mx); s += e[j]; }
  float inv = 1.0f / s;
#pragma unroll
  for (int j = 0; j < 12; ++j) p[j] = e[j] * inv;
}

// ---------------- deformable bilinear sampling ----------------
// one wave per (b,q,h); lane = channel d (DH=32)
__global__ __launch_bounds__(256) void deform_sample(
    const float* __restrict__ val,   // [M, 256] head-major columns
    const float* __restrict__ off,   // [M, 192]
    const float* __restrict__ aw,    // [M, 128] (first 96 valid)
    float* __restrict__ out)         // [M, 256]
{
  int wv = blockIdx.x * 8 + (threadIdx.x >> 5);
  int d  = threadIdx.x & 31;
  if (wv >= M * NH) return;
  int h = wv % NH;
  int t = wv / NH;           // b*Q + q
  int q = t % Q;
  int b = t / Q;

  float refx, refy;
  {
    int w0, qi;
    if (q < 4096)      { w0 = 64; qi = q; }
    else if (q < 5120) { w0 = 32; qi = q - 4096; }
    else               { w0 = 16; qi = q - 5120; }
    int r = qi / w0, c = qi % w0;
    float invw = 1.0f / (float)w0;
    refx = (c + 0.5f) * invw;
    refy = (r + 0.5f) * invw;   // square levels
  }

  const float* offp = off + (size_t)t * (NH * NL * NP * 2) + h * (NL * NP * 2);
  const float* awp  = aw  + (size_t)t * NATTP + h * (NL * NP);

  const int starts[NL] = {0, 4096, 5120};
  const int sizes [NL] = {64, 32, 16};

  float acc = 0.0f;
#pragma unroll
  for (int l = 0; l < NL; ++l) {
    const int wl = sizes[l], hl = sizes[l], sl = starts[l];
    const float* vb = val + ((size_t)(b * Q + sl)) * D + h * DH + d;
#pragma unroll
    for (int p = 0; p < NP; ++p) {
      float ox = offp[l * 8 + p * 2 + 0];
      float oy = offp[l * 8 + p * 2 + 1];
      float a  = awp [l * 4 + p];
      float px = refx * (float)wl + ox - 0.5f;
      float py = refy * (float)hl + oy - 0.5f;
      float x0f = floorf(px), y0f = floorf(py);
      float lx = px - x0f, ly = py - y0f;
      int x0 = (int)x0f, y0 = (int)y0f;
      float w00 = (1.f - lx) * (1.f - ly) * a;
      float w10 = lx * (1.f - ly) * a;
      float w01 = (1.f - lx) * ly * a;
      float w11 = lx * ly * a;
      bool vx0 = (unsigned)x0       < (unsigned)wl;
      bool vx1 = (unsigned)(x0 + 1) < (unsigned)wl;
      bool vy0 = (unsigned)y0       < (unsigned)hl;
      bool vy1 = (unsigned)(y0 + 1) < (unsigned)hl;
      if (vx0 && vy0) acc += w00 * vb[(size_t)(y0 * wl + x0) * D];
      if (vx1 && vy0) acc += w10 * vb[(size_t)(y0 * wl + x0 + 1) * D];
      if (vx0 && vy1) acc += w01 * vb[(size_t)((y0 + 1) * wl + x0) * D];
      if (vx1 && vy1) acc += w11 * vb[(size_t)((y0 + 1) * wl + x0 + 1) * D];
    }
  }
  out[(size_t)t * D + h * DH + d] = acc;
}

// ---------------- fused residual + LayerNorm (wave per row) ----------------
__global__ __launch_bounds__(256) void resid_ln(
    const float* __restrict__ x, const float* __restrict__ delta,
    const float* __restrict__ g, const float* __restrict__ b,
    float* __restrict__ xout, unsigned short* __restrict__ xbout, int rows)
{
  int row  = blockIdx.x * 8 + (threadIdx.x >> 5);
  int lane = threadIdx.x & 31;
  if (row >= rows) return;
  const float* xr = x     + (size_t)row * D;
  const float* dr = delta + (size_t)row * D;
  float v[8];
  float s = 0.f;
#pragma unroll
  for (int r = 0; r < 8; ++r) {
    v[r] = xr[lane + r * 32] + dr[lane + r * 32];
    s += v[r];
  }
#pragma unroll
  for (int m = 16; m > 0; m >>= 1) s += __shfl_xor(s, m, 32);
  float mean = s * (1.0f / D);
  float vs = 0.f;
#pragma unroll
  for (int r = 0; r < 8; ++r) { float dv = v[r] - mean; vs += dv * dv; }
#pragma unroll
  for (int m = 16; m > 0; m >>= 1) vs += __shfl_xor(vs, m, 32);
  float rstd = rsqrtf(vs * (1.0f / D) + 1e-5f);
#pragma unroll
  for (int r = 0; r < 8; ++r) {
    int c = lane + r * 32;
    float y = (v[r] - mean) * rstd * g[c] + b[c];
    xout [(size_t)row * D + c] = y;
    xbout[(size_t)row * D + c] = f2bf(y);
  }
}

// ---------------- host side ----------------
static inline int ceildiv(int a, int b) { return (a + b - 1) / b; }

extern "C" void kernel_launch(void* const* d_in, const int* in_sizes, int n_in,
                              void* d_out, int out_size, void* d_ws, size_t ws_size,
                              hipStream_t stream)
{
  (void)in_sizes; (void)n_in; (void)out_size; (void)ws_size;
  const float* src0 = (const float*)d_in[0];
  const float* pos0 = (const float*)d_in[1];
  const float* src1 = (const float*)d_in[2];
  const float* pos1 = (const float*)d_in[3];
  const float* src2 = (const float*)d_in[4];
  const float* pos2 = (const float*)d_in[5];
  const float* level_embed = (const float*)d_in[6];
  const float* W_off  = (const float*)d_in[7];
  const float* b_off  = (const float*)d_in[8];
  const float* W_attn = (const float*)d_in[9];
  const float* b_attn = (const float*)d_in[10];
  const float* W_val  = (const float*)d_in[11];
  const float* b_val  = (const float*)d_in[12];
  const float* W_out  = (const float*)d_in[13];
  const float* b_out  = (const float*)d_in[14];
  const float* ln1_g  = (const float*)d_in[15];
  const float* ln1_b  = (const float*)d_in[16];
  const float* W_ff1  = (const float*)d_in[17];
  const float* b_ff1  = (const float*)d_in[18];
  const float* W_ff2  = (const float*)d_in[19];
  const float* b_ff2  = (const float*)d_in[20];
  const float* ln2_g  = (const float*)d_in[21];
  const float* ln2_b  = (const float*)d_in[22];

  // -------- workspace carving --------
  char* wsb = (char*)d_ws;
  size_t o = 0;
  auto carve = [&](size_t bytes) -> char* {
    char* p = wsb + o;
    o += (bytes + 255) & ~(size_t)255;
    return p;
  };
  const size_t SZ_XF = (size_t)M * D * sizeof(float);
  float*          xf    = (float*)carve(SZ_XF);
  float*          posf  = (float*)carve(SZ_XF);
  unsigned short* xb    = (unsigned short*)carve((size_t)M * D * 2);
  unsigned short* qb    = (unsigned short*)carve((size_t)M * D * 2);
  float*          delta = (float*)carve(SZ_XF);
  char*           scr   = carve((size_t)M * DFF * 2);   // samp f32 / ffh bf16
  float*          samp  = (float*)scr;
  unsigned short* ffh   = (unsigned short*)scr;
  float*          offb  = (float*)carve((size_t)M * (NH*NL*NP*2) * 4);
  float*          awb   = (float*)carve((size_t)M * NATTP * 4);
  float*          valb  = (float*)carve(SZ_XF);
  unsigned short* wt_off  = (unsigned short*)carve((size_t)NLAYERS * 192   * D   * 2);
  unsigned short* wt_attn = (unsigned short*)carve((size_t)NLAYERS * NATTP * D   * 2);
  unsigned short* wt_val  = (unsigned short*)carve((size_t)NLAYERS * D     * D   * 2);
  unsigned short* wt_out  = (unsigned short*)carve((size_t)NLAYERS * D     * D   * 2);
  unsigned short* wt_ff1  = (unsigned short*)carve((size_t)NLAYERS * DFF   * D   * 2);
  unsigned short* wt_ff2  = (unsigned short*)carve((size_t)NLAYERS * D     * DFF * 2);
  float*          bpad    = (float*)carve((size_t)NLAYERS * NATTP * 4);

  // -------- weight transpose + bf16 conversion (every call; deterministic) --------
  auto cvtw = [&](const float* in, unsigned short* out, int K, int N, int Npad) {
    int n = NLAYERS * Npad * K;
    cvt_w_transpose<<<ceildiv(n, 256), 256, 0, stream>>>(in, out, K, N, Npad, NLAYERS);
  };
  cvtw(W_off,  wt_off,  D,   192, 192);
  cvtw(W_attn, wt_attn, D,   NATT, NATTP);
  cvtw(W_val,  wt_val,  D,   D,   D);
  cvtw(W_out,  wt_out,  D,   D,   D);
  cvtw(W_ff1,  wt_ff1,  D,   DFF, DFF);
  cvtw(W_ff2,  wt_ff2,  DFF, D,   D);
  pad_bias_attn<<<ceildiv(NLAYERS * NATTP, 256), 256, 0, stream>>>(b_attn, bpad);

  // -------- flatten inputs --------
  flatten_kernel<<<ceildiv(M * D, 256), 256, 0, stream>>>(
      src0, pos0, src1, pos1, src2, pos2, level_embed, xf, posf);

  // -------- GEMM launcher --------
  auto gemm = [&](const unsigned short* A, const unsigned short* Bt, const float* bias,
                  float* Cf, unsigned short* Cb, int N, int K, bool relu, bool outbf) {
    dim3 grid(M / GBM, N / GBN);
    if (outbf) {
      if (relu) gemm_wmma<true,  true ><<<grid, 256, 0, stream>>>(A, Bt, bias, Cf, Cb, M, N, K);
      else      gemm_wmma<false, true ><<<grid, 256, 0, stream>>>(A, Bt, bias, Cf, Cb, M, N, K);
    } else {
      if (relu) gemm_wmma<true,  false><<<grid, 256, 0, stream>>>(A, Bt, bias, Cf, Cb, M, N, K);
      else      gemm_wmma<false, false><<<grid, 256, 0, stream>>>(A, Bt, bias, Cf, Cb, M, N, K);
    }
  };

  // -------- encoder layers --------
  for (int i = 0; i < NLAYERS; ++i) {
    prep_qx<<<ceildiv(M * D, 256), 256, 0, stream>>>(xf, posf, qb, xb, M * D);

    gemm(qb, wt_off  + (size_t)i * 192   * D, b_off + (size_t)i * 192, offb, nullptr, 192,   D, false, false);
    gemm(qb, wt_attn + (size_t)i * NATTP * D, bpad  + (size_t)i * NATTP, awb, nullptr, NATTP, D, false, false);
    gemm(xb, wt_val  + (size_t)i * D     * D, b_val + (size_t)i * D,   valb, nullptr, D,     D, false, false);

    softmax12<<<ceildiv(M * NH, 256), 256, 0, stream>>>(awb);

    deform_sample<<<ceildiv(M * NH, 8), 256, 0, stream>>>(valb, offb, awb, samp);

    cvt_f32_to_bf16<<<ceildiv(M * D, 256), 256, 0, stream>>>(samp, qb, M * D);
    gemm(qb, wt_out + (size_t)i * D * D, b_out + (size_t)i * D, delta, nullptr, D, D, false, false);

    resid_ln<<<ceildiv(M, 8), 256, 0, stream>>>(
        xf, delta, ln1_g + (size_t)i * D, ln1_b + (size_t)i * D, xf, xb, M);

    gemm(xb,  wt_ff1 + (size_t)i * DFF * D, b_ff1 + (size_t)i * DFF, nullptr, ffh, DFF, D,   true,  true);
    gemm(ffh, wt_ff2 + (size_t)i * D * DFF, b_ff2 + (size_t)i * D,   delta, nullptr, D,   DFF, false, false);

    resid_ln<<<ceildiv(M, 8), 256, 0, stream>>>(
        xf, delta, ln2_g + (size_t)i * D, ln2_b + (size_t)i * D, xf, xb, M);
  }

  hipMemcpyAsync(d_out, xf, SZ_XF, hipMemcpyDeviceToDevice, stream);
}
